// KnowledgeConsistentAttention_13434657702219
// MI455X (gfx1250) — compile-verified, matrix operand order checked by reading, not run on hardware
//
#include <hip/hip_runtime.h>

typedef __attribute__((ext_vector_type(16))) _Float16 v16h;
typedef __attribute__((ext_vector_type(8)))  _Float16 v8h;
typedef __attribute__((ext_vector_type(8)))  float    v8f;
typedef __attribute__((ext_vector_type(4)))  float    v4f;
typedef __attribute__((ext_vector_type(4)))  unsigned v4u;
typedef __attribute__((ext_vector_type(4)))  int      v4i;
typedef __attribute__((ext_vector_type(8)))  int      v8i;

#define BZ   4
#define HH   64
#define WW   64
#define NQ   4096      /* 64*64 */
#define NC   256
#define EPSF 1e-7f
#define NCHUNK (NQ / 32)

// ---------------------------------------------------------------------------
// Prep 1: Kh[b*q][c] = (F+eps)/||F+eps||  (f16), and KT[b][c][q] = Kh^T (f16)
// ---------------------------------------------------------------------------
__global__ __launch_bounds__(256) void prep_norm(const float* __restrict__ F,
                                                 _Float16* __restrict__ Kh,
                                                 _Float16* __restrict__ KT) {
  int wave = threadIdx.x >> 5;
  int lane = threadIdx.x & 31;
  int row  = blockIdx.x * 8 + wave;                 // [0, BZ*NQ)
  const float* src = F + (size_t)row * NC + lane * 8;
  v4f a = *(const v4f*)(src);
  v4f b = *(const v4f*)(src + 4);
  float v[8];
  float s = 0.f;
#pragma unroll
  for (int i = 0; i < 4; ++i) { v[i]     = a[i] + EPSF; s += v[i]     * v[i];     }
#pragma unroll
  for (int i = 0; i < 4; ++i) { v[4 + i] = b[i] + EPSF; s += v[4 + i] * v[4 + i]; }
#pragma unroll
  for (int off = 16; off >= 1; off >>= 1) s += __shfl_xor(s, off, 32);
  float rn = rsqrtf(s);
  int bb = row >> 12;
  int q  = row & (NQ - 1);
  _Float16* kh = Kh + (size_t)row * NC + lane * 8;
  _Float16* kt = KT + (size_t)bb * NC * NQ + q;
#pragma unroll
  for (int i = 0; i < 8; ++i) {
    _Float16 hv = (_Float16)(v[i] * rn);
    kh[i] = hv;
    kt[(size_t)(lane * 8 + i) * NQ] = hv;
  }
}

// ---------------------------------------------------------------------------
// Prep 2: G[b][p][c] = (9/cnt(p)) * sum_{3x3 SAME} F[b][n][c]   (f16)
// ---------------------------------------------------------------------------
__global__ __launch_bounds__(256) void prep_pool(const float* __restrict__ F,
                                                 _Float16* __restrict__ G) {
  int idx = blockIdx.x * blockDim.x + threadIdx.x;  // [0, BZ*NQ*NC/4)
  int c4  = (idx & (NC / 4 - 1)) * 4;
  int p   = (idx >> 6) & (NQ - 1);
  int b   = idx >> 18;
  int i = p >> 6, j = p & 63;
  float s0 = 0.f, s1 = 0.f, s2 = 0.f, s3 = 0.f;
  int cnt = 0;
#pragma unroll
  for (int di = -1; di <= 1; ++di) {
    int ii = i + di;
    if (ii < 0 || ii >= HH) continue;
#pragma unroll
    for (int dj = -1; dj <= 1; ++dj) {
      int jj = j + dj;
      if (jj < 0 || jj >= WW) continue;
      const float* src = F + (((size_t)b * NQ + ii * WW + jj) * NC) + c4;
      v4f x = *(const v4f*)src;
      s0 += x[0]; s1 += x[1]; s2 += x[2]; s3 += x[3];
      ++cnt;
    }
  }
  float sc = 9.0f / (float)cnt;
  _Float16* g = G + ((size_t)b * NQ + p) * NC + c4;
  g[0] = (_Float16)(s0 * sc);
  g[1] = (_Float16)(s1 * sc);
  g[2] = (_Float16)(s2 * sc);
  g[3] = (_Float16)(s3 * sc);
}

// ---------------------------------------------------------------------------
// WMMA 16-bit A/B fragment loader from a row-major matrix (global or LDS):
//   lanes 0-15 : row = lane,    halves k = {0..7, 16..23}
//   lanes 16-31: row = lane-16, halves k = {8..15, 24..31}
// ---------------------------------------------------------------------------
__device__ __forceinline__ v16h ld_frag(const _Float16* p, int stride, int lane) {
  int hs = (lane & 16) ? 8 : 0;
  const _Float16* r = p + (size_t)(lane & 15) * stride + hs;
  v8h lo = *(const v8h*)(r);
  v8h hi = *(const v8h*)(r + 16);
  v16h o;
#pragma unroll
  for (int i = 0; i < 8; ++i) { o[i] = lo[i]; o[i + 8] = hi[i]; }
  return o;
}

// ---------------------------------------------------------------------------
// Tensor Data Mover helpers (D# per CDNA5 ISA)
// ---------------------------------------------------------------------------
__device__ __forceinline__ void tdm_issue(v4u g0, v8i g1) {
  v4i z4 = {};
#if defined(__clang_major__) && (__clang_major__ >= 23)
  v8i z8 = {};
  __builtin_amdgcn_tensor_load_to_lds(g0, g1, z4, z4, z8, 0);
#else
  __builtin_amdgcn_tensor_load_to_lds(g0, g1, z4, z4, 0);
#endif
}

__device__ __forceinline__ v4u tdm_g0(unsigned lds_addr, const void* gptr) {
  unsigned long long ga = (unsigned long long)(uintptr_t)gptr;
  v4u g0;
  g0.x = 1u;                                              // count = 1
  g0.y = lds_addr;                                        // LDS byte address
  g0.z = (unsigned)ga;                                    // global_addr[31:0]
  g0.w = ((unsigned)(ga >> 32) & 0x01FFFFFFu) | (2u << 30); // addr[56:32], type=2
  return g0;
}

// 1-D contiguous tile: tile_elems halves starting at global_addr
__device__ __forceinline__ v8i tdm_g1_1d(unsigned tile_elems, unsigned tensor_elems) {
  v8i d = {};
  d[0] = (int)(1u << 16);                                  // data_size = 2 bytes
  d[1] = (int)((tensor_elems & 0xffffu) << 16);            // tensor_dim0[15:0] @48
  d[2] = (int)(((tensor_elems >> 16) & 0xffffu)            // tensor_dim0[31:16]
               | (1u << 16));                              // tensor_dim1 = 1 @80
  d[3] = (int)(tile_elems << 16);                          // tile_dim0 @112
  d[4] = 1;                                                // tile_dim1 = 1 @128
  d[5] = (int)tensor_elems;                                // tensor_dim0_stride @160
  return d;
}

// 2-D tile: tile0 x tile1 from a td0 x td1 tensor with row stride stride0 (halves)
__device__ __forceinline__ v8i tdm_g1_2d(unsigned tile0, unsigned tile1,
                                         unsigned td0, unsigned td1, unsigned stride0) {
  v8i d = {};
  d[0] = (int)(1u << 16);                                  // data_size = 2 bytes
  d[1] = (int)((td0 & 0xffffu) << 16);                     // tensor_dim0[15:0] @48
  d[2] = (int)(((td0 >> 16) & 0xffffu) | ((td1 & 0xffffu) << 16)); // td0 hi | td1 lo
  d[3] = (int)(((td1 >> 16) & 0xffffu) | (tile0 << 16));   // td1 hi | tile_dim0 @112
  d[4] = (int)(tile1 & 0xffffu);                           // tile_dim1 @128
  d[5] = (int)stride0;                                     // tensor_dim0_stride @160
  return d;
}

// ---------------------------------------------------------------------------
// Fused flash attention: out = softmax(G . Kh^T) . Kh
// TDM double-buffers the 32-key Kh/KT tiles into LDS; 4 waves share them.
// ---------------------------------------------------------------------------
__global__ __launch_bounds__(128) void flash_attn(const _Float16* __restrict__ G,
                                                  const _Float16* __restrict__ Kh,
                                                  const _Float16* __restrict__ KT,
                                                  float* __restrict__ Out) {
  __shared__ __align__(16) _Float16 KhT[2][32 * NC];   // [keys][c]   16 KB x2
  __shared__ __align__(16) _Float16 KTT[2][NC * 32];   // [c][keys]   16 KB x2
  __shared__ __align__(16) _Float16 Plds[4][16 * 32];  // per-wave P bounce

  int wave = threadIdx.x >> 5;
  int lane = threadIdx.x & 31;
  int b    = blockIdx.y;
  int q0   = blockIdx.x * 64 + wave * 16;

  const _Float16* Gb  = G  + (size_t)b * NQ * NC;
  const _Float16* Khb = Kh + (size_t)b * NQ * NC;
  const _Float16* KTb = KT + (size_t)b * NC * NQ;

  // Resident Q fragments: 8 chunks of 16x32 f16
  v16h Q[8];
#pragma unroll
  for (int cc = 0; cc < 8; ++cc)
    Q[cc] = ld_frag(Gb + (size_t)q0 * NC + cc * 32, NC, lane);

  v8f zero = {};
  v8f O[16];
#pragma unroll
  for (int t = 0; t < 16; ++t) O[t] = zero;
  float m[8], l[8];
#pragma unroll
  for (int r = 0; r < 8; ++r) { m[r] = -3.0e38f; l[r] = 0.f; }

  _Float16* P = Plds[wave];
  int col     = lane & 15;
  int rowbase = (lane & 16) ? 8 : 0;

  // Prologue: wave 0 issues TDM for chunks 0 and 1 (2 descriptors each)
  if (wave == 0) {
#pragma unroll
    for (int c0 = 0; c0 < 2; ++c0) {
      tdm_issue(tdm_g0((unsigned)(uintptr_t)&KhT[c0][0], Khb + (size_t)(c0 * 32) * NC),
                tdm_g1_1d(32 * NC, NQ * NC));
      tdm_issue(tdm_g0((unsigned)(uintptr_t)&KTT[c0][0], KTb + c0 * 32),
                tdm_g1_2d(32, NC, NQ, NC, NQ));
    }
  }

  for (int c = 0; c < NCHUNK; ++c) {
    int buf = c & 1;
    if (wave == 0) {
      if (c < NCHUNK - 1) __builtin_amdgcn_s_wait_tensorcnt(2);  // chunk c landed
      else                __builtin_amdgcn_s_wait_tensorcnt(0);
    }
    __syncthreads();   // LDS tiles for chunk c visible to all waves

    // ---- S = Q . K^T for two 16-key sub-tiles (8 k-chunks of 32 channels) ----
    v8f S0 = zero, S1 = zero;
#pragma unroll
    for (int cc = 0; cc < 8; ++cc) {
      v16h b0 = ld_frag(&KhT[buf][cc * 32],            NC, lane);
      v16h b1 = ld_frag(&KhT[buf][16 * NC + cc * 32],  NC, lane);
      S0 = __builtin_amdgcn_wmma_f32_16x16x32_f16(false, Q[cc], false, b0,
                                                  (short)0, S0, false, false);
      S1 = __builtin_amdgcn_wmma_f32_16x16x32_f16(false, Q[cc], false, b1,
                                                  (short)0, S1, false, false);
    }

    // ---- online softmax; C-layout: VGPR r = row (rowbase+r), lane%16 = col ----
#pragma unroll
    for (int r = 0; r < 8; ++r) {
      float cm = fmaxf(S0[r], S1[r]);
#pragma unroll
      for (int off = 1; off < 16; off <<= 1) cm = fmaxf(cm, __shfl_xor(cm, off, 32));
      float nm    = fmaxf(m[r], cm);
      float alpha = __expf(m[r] - nm);
      m[r] = nm;
      float p0 = __expf(S0[r] - nm);
      float p1 = __expf(S1[r] - nm);
      float ps = p0 + p1;
#pragma unroll
      for (int off = 1; off < 16; off <<= 1) ps += __shfl_xor(ps, off, 32);
      l[r] = l[r] * alpha + ps;
#pragma unroll
      for (int t = 0; t < 16; ++t) O[t][r] *= alpha;
      P[(rowbase + r) * 32 + col]      = (_Float16)p0;
      P[(rowbase + r) * 32 + 16 + col] = (_Float16)p1;
    }

    // ---- O += P . V  (V tiles read transposed from LDS KTT) ----
    v16h Pa = ld_frag(P, 32, lane);
#pragma unroll
    for (int t = 0; t < 16; ++t) {
      v16h bv = ld_frag(&KTT[buf][(t * 16) * 32], 32, lane);
      O[t] = __builtin_amdgcn_wmma_f32_16x16x32_f16(false, Pa, false, bv,
                                                    (short)0, O[t], false, false);
    }

    __syncthreads();   // all waves done reading buf before it is overwritten
    if (wave == 0 && (c + 2) < NCHUNK) {
      int key0 = (c + 2) * 32;
      tdm_issue(tdm_g0((unsigned)(uintptr_t)&KhT[buf][0], Khb + (size_t)key0 * NC),
                tdm_g1_1d(32 * NC, NQ * NC));
      tdm_issue(tdm_g0((unsigned)(uintptr_t)&KTT[buf][0], KTb + key0),
                tdm_g1_2d(32, NC, NQ, NC, NQ));
    }
  }

  // ---- normalize and store fp32 output ----
  float* ob = Out + (size_t)b * NQ * NC;
#pragma unroll
  for (int r = 0; r < 8; ++r) {
    float inv   = 1.0f / l[r];
    size_t base = (size_t)(q0 + rowbase + r) * NC;
#pragma unroll
    for (int t = 0; t < 16; ++t)
      ob[base + t * 16 + col] = O[t][r] * inv;
  }
}

// ---------------------------------------------------------------------------
extern "C" void kernel_launch(void* const* d_in, const int* in_sizes, int n_in,
                              void* d_out, int out_size, void* d_ws, size_t ws_size,
                              hipStream_t stream) {
  const float* F = (const float*)d_in[0];     // foreground [4,64,64,256] fp32
  // d_in[1] (masks) does not affect the output — intentionally unused.
  float* out = (float*)d_out;                 // [4,64,64,256] fp32

  const size_t HALF_MAT = (size_t)BZ * NQ * NC * sizeof(_Float16); // 8 MB
  char* ws = (char*)d_ws;
  _Float16* Kh = (_Float16*)(ws);                 // [b][q][c]
  _Float16* KT = (_Float16*)(ws + HALF_MAT);      // [b][c][q]
  _Float16* G  = (_Float16*)(ws + 2 * HALF_MAT);  // [b][p][c]

  prep_norm<<<dim3(BZ * NQ / 8), dim3(256), 0, stream>>>(F, Kh, KT);
  prep_pool<<<dim3(BZ * NQ * (NC / 4) / 256), dim3(256), 0, stream>>>(F, G);
  flash_attn<<<dim3(NQ / 64, BZ), dim3(128), 0, stream>>>(G, Kh, KT, out);
}
